// DeepARPoint_50010599195066
// MI455X (gfx1250) — compile-verified, matrix operand order checked by reading, not run on hardware
//
#include <hip/hip_runtime.h>

typedef __attribute__((ext_vector_type(16))) __bf16 v16bf;
typedef __attribute__((ext_vector_type(8)))  float  v8f;

#define BATCH    128
#define CTXLEN   720
#define HORLEN   168
#define TSTEPS   887          // CTX + HOR - 1 output steps
#define HID      512
#define G4       2048         // 4*HID
#define K0       544          // 32 (padded x) + 512 (h0)
#define KT0      17           // K0/32
#define K1       1024         // 512 (h0) + 512 (h1)
#define KT1      32           // K1/32
#define NT       128          // N tiles (2048/16)
#define MT       8            // M tiles (128/16)
#define NBLK     128
#define NTHR     256
#define T0LEN    916          // CTX + HOR + MAX_LAG
#define MAXLAG   28

__device__ __constant__ int c_lags[10] = {1,2,3,4,5,6,7,14,21,28};

__device__ __forceinline__ unsigned short f2bf(float f) {
    unsigned u = __float_as_uint(f);
    u = (u + 0x7FFFu + ((u >> 16) & 1u)) >> 16;   // RNE
    return (unsigned short)u;
}
__device__ __forceinline__ float sigm(float x) { return 1.0f / (1.0f + expf(-x)); }

// ---------------- grid-wide ticket barrier -------------------------------
__device__ __forceinline__ void gbar(unsigned* cnt) {
    __syncthreads();
    if (threadIdx.x == 0) {
        __threadfence();
        unsigned ticket = atomicAdd(cnt, 1u);
        unsigned target = (ticket / NBLK + 1u) * NBLK;
        while (*(volatile unsigned*)cnt < target) {
            __builtin_amdgcn_s_sleep(2);
        }
        __threadfence();
    }
    __syncthreads();
}

// ---------------- WMMA GEMM phase: one 16x16 tile per wave ----------------
// A: row-major bf16 [128][SK]  (per-lane 16 contiguous K, k = 16*(lane/16) + s; Wp uses the same)
// Wp: pre-swizzled bf16 B tiles: [(ntile*KT + k)*32 + lane][16]
template<int KT, int SK>
__device__ __forceinline__ void gemm_tiles(const unsigned short* __restrict__ A,
                                           const unsigned short* __restrict__ Wp,
                                           float* __restrict__ Gout,
                                           int wgid, int lane) {
    const int mtile = wgid & (MT - 1);
    const int ntile = wgid >> 3;
    const unsigned short* arow = A + (size_t)(mtile * 16 + (lane & 15)) * SK + (lane >> 4) * 16;
    const unsigned short* bp   = Wp + (size_t)ntile * KT * 512 + (size_t)lane * 16;
    v8f acc = {};
#pragma unroll
    for (int k = 0; k < KT; ++k) {
        v16bf a = *(const v16bf*)(arow + k * 32);
        v16bf b = *(const v16bf*)(bp + k * 512);
        if (k + 1 < KT) __builtin_prefetch(bp + (k + 1) * 512, 0, 1);
        acc = __builtin_amdgcn_wmma_f32_16x16x32_bf16(false, a, false, b,
                                                      (short)0, acc, false, false);
    }
    float* gp = Gout + (size_t)(mtile * 16 + 8 * (lane >> 4)) * G4 + ntile * 16 + (lane & 15);
#pragma unroll
    for (int r = 0; r < 8; ++r) gp[(size_t)r * G4] = acc[r];
}

// ---------------- prep: per-batch scale / logscale / lag buffer -----------
__global__ void prep_scale_k(const float* __restrict__ X,
                             const unsigned char* __restrict__ mask,
                             float* __restrict__ scale, float* __restrict__ logscale,
                             float* __restrict__ lagbuf) {
    const int b = blockIdx.x, tid = threadIdx.x;
    __shared__ float s_sum[NTHR], s_cnt[NTHR];
    __shared__ float s_scale;
    float sum = 0.f, cnt = 0.f;
    for (int i = tid; i < CTXLEN; i += NTHR) {
        float m = mask[b * T0LEN + MAXLAG + i] ? 1.f : 0.f;
        float v = X[(size_t)(b * T0LEN + MAXLAG + i) * 2];
        sum += fabsf(v) * m; cnt += m;
    }
    s_sum[tid] = sum; s_cnt[tid] = cnt; __syncthreads();
    for (int s = NTHR / 2; s > 0; s >>= 1) {
        if (tid < s) { s_sum[tid] += s_sum[tid + s]; s_cnt[tid] += s_cnt[tid + s]; }
        __syncthreads();
    }
    if (tid == 0) {
        float sc = s_sum[0] / fmaxf(s_cnt[0], 1.f);
        sc = fmaxf(sc, 1e-10f);
        scale[b] = sc; logscale[b] = logf(sc); s_scale = sc;
    }
    __syncthreads();
    float sc = s_scale;
    for (int j = tid; j < MAXLAG; j += NTHR)   // past[b, j] = X[b, 747 - j, 0]
        lagbuf[b * MAXLAG + j] = X[(size_t)(b * T0LEN + (T0LEN - HORLEN - 1) - j) * 2] / sc;
}

// ---------------- prep: static step features (bf16, K padded to 32) -------
__global__ void prep_feat_k(const float* __restrict__ X, const float* __restrict__ emb,
                            const float* __restrict__ scale, const float* __restrict__ logscale,
                            unsigned short* __restrict__ xq) {
    int idx = blockIdx.x * blockDim.x + threadIdx.x;
    if (idx >= TSTEPS * BATCH) return;
    const int t = idx / BATCH, b = idx % BATCH;
    const float sc = scale[b];
    unsigned short* dst = xq + (size_t)idx * 32;
    const int base = b * T0LEN + MAXLAG + t;
    dst[0] = (t < CTXLEN) ? f2bf(X[(size_t)base * 2] / sc) : (unsigned short)0;
#pragma unroll
    for (int l = 0; l < 10; ++l)
        dst[1 + l] = (t < CTXLEN)
            ? f2bf(X[(size_t)(base - c_lags[l]) * 2] / sc) : (unsigned short)0;
    dst[11] = f2bf(logscale[b]);
    int cat = (int)X[(size_t)base * 2 + 1];
    cat = cat < 0 ? 0 : (cat > 11 ? 11 : cat);
#pragma unroll
    for (int e = 0; e < 5; ++e) dst[12 + e] = f2bf(emb[cat * 5 + e]);
#pragma unroll
    for (int k = 17; k < 32; ++k) dst[k] = 0;
}

// ---------------- prep: pack concat weights into WMMA-B tile order --------
__global__ void pack_w_k(const float* __restrict__ WA, int ldA, int KvalidA,
                         const float* __restrict__ WB, int Ksplit, int ktiles,
                         unsigned short* __restrict__ outp) {
    int idx = blockIdx.x * blockDim.x + threadIdx.x;
    if (idx >= NT * ktiles * 32) return;
    const int lane  = idx & 31;
    const int k     = (idx >> 5) % ktiles;
    const int ntile = (idx >> 5) / ktiles;
    const int N     = ntile * 16 + (lane & 15);
    const int Kbase = k * 32 + (lane >> 4) * 16;
    unsigned short* dst = outp + ((size_t)(ntile * ktiles + k) * 32 + lane) * 16;
#pragma unroll
    for (int e = 0; e < 16; ++e) {
        int K = Kbase + e;
        float v = (K < Ksplit) ? ((K < KvalidA) ? WA[(size_t)N * ldA + K] : 0.f)
                               : WB[(size_t)N * 512 + (K - Ksplit)];
        dst[e] = f2bf(v);
    }
}

__global__ void prep_bias_k(const float* a, const float* b, const float* c, const float* d,
                            float* bias0, float* bias1) {
    int i = blockIdx.x * blockDim.x + threadIdx.x;
    if (i < G4) bias0[i] = a[i] + b[i];
    else if (i < 2 * G4) { int j = i - G4; bias1[j] = c[j] + d[j]; }
}

__global__ void prep_state_k(float* c0, float* c1, unsigned short* hA0,
                             unsigned short* hA1, const unsigned short* xq) {
    int i = blockIdx.x * blockDim.x + threadIdx.x;
    int stride = gridDim.x * blockDim.x;
    for (int e = i; e < BATCH * HID; e += stride) { c0[e] = 0.f; c1[e] = 0.f; }
    for (int e = i; e < BATCH * K1; e += stride) hA1[e] = 0;
    for (int e = i; e < BATCH * K0; e += stride) {
        int b = e / K0, k = e % K0;
        hA0[e] = (k < 32) ? xq[(size_t)b * 32 + k] : (unsigned short)0;
    }
}

// ---------------- persistent recurrent kernel -----------------------------
__global__ void __launch_bounds__(NTHR)
lstm_persistent(const unsigned short* __restrict__ wcat0,
                const unsigned short* __restrict__ wcat1,
                const float* __restrict__ bias0, const float* __restrict__ bias1,
                const unsigned short* __restrict__ xq,
                unsigned short* __restrict__ hA0, unsigned short* __restrict__ hA1,
                float* __restrict__ g0, float* __restrict__ g1,
                float* __restrict__ c0, float* __restrict__ c1,
                float* __restrict__ h1f, float* __restrict__ lagbuf,
                const float* __restrict__ scale,
                const float* __restrict__ W_head, const float* __restrict__ b_head,
                float* __restrict__ yout, unsigned* __restrict__ barrier) {
    const int tid  = threadIdx.x;
    const int lane = tid & 31;
    const int wgid = blockIdx.x * (NTHR / 32) + (tid >> 5);   // 0..1023
    const int gtid = blockIdx.x * NTHR + tid;                 // 0..32767
    const float bhead = b_head[0];

    for (int t = 0; t < TSTEPS; ++t) {
        // ---- phase A: layer0 gates  g0 = [x|h0] @ Wcat0 -----------------
        gemm_tiles<KT0, K0>(hA0, wcat0, g0, wgid, lane);
        gbar(barrier);

        // ---- phase B: layer0 cell update --------------------------------
        for (int e = gtid; e < BATCH * HID; e += NBLK * NTHR) {
            const int b = e >> 9, j = e & (HID - 1);
            const float* gr = g0 + (size_t)b * G4;
            float gi = gr[j]          + bias0[j];
            float gf = gr[512 + j]    + bias0[512 + j];
            float gg = gr[1024 + j]   + bias0[1024 + j];
            float go = gr[1536 + j]   + bias0[1536 + j];
            float cn = sigm(gf) * c0[e] + sigm(gi) * tanhf(gg);
            float h  = sigm(go) * tanhf(cn);
            c0[e] = cn;
            unsigned short hb = f2bf(h);
            hA1[(size_t)b * K1 + j]      = hb;   // layer1 A (h0 half)
            hA0[(size_t)b * K0 + 32 + j] = hb;   // next-step layer0 A (h0 half)
        }
        gbar(barrier);

        // ---- phase C: layer1 gates  g1 = [h0|h1] @ Wcat1 ----------------
        gemm_tiles<KT1, K1>(hA1, wcat1, g1, wgid, lane);
        gbar(barrier);

        // ---- phase D: layer1 cell update --------------------------------
        for (int e = gtid; e < BATCH * HID; e += NBLK * NTHR) {
            const int b = e >> 9, j = e & (HID - 1);
            const float* gr = g1 + (size_t)b * G4;
            float gi = gr[j]          + bias1[j];
            float gf = gr[512 + j]    + bias1[512 + j];
            float gg = gr[1024 + j]   + bias1[1024 + j];
            float go = gr[1536 + j]   + bias1[1536 + j];
            float cn = sigm(gf) * c1[e] + sigm(gi) * tanhf(gg);
            float h  = sigm(go) * tanhf(cn);
            c1[e]  = cn;
            h1f[e] = h;
            hA1[(size_t)b * K1 + 512 + j] = f2bf(h);
        }
        gbar(barrier);

        // ---- phase E: head projection + next-step input staging ---------
        if (wgid < BATCH) {
            const int b = wgid;
            float p = 0.f;
            for (int j = lane; j < HID; j += 32) p += h1f[(size_t)b * HID + j] * W_head[j];
#pragma unroll
            for (int off = 16; off > 0; off >>= 1) p += __shfl_xor(p, off, 32);
            const float y = p + bhead;                       // normalized prediction
            if (lane == 0) yout[(size_t)b * TSTEPS + t] = y * scale[b];
            if (t < TSTEPS - 1) {
                const int tn = t + 1;
                unsigned short v = xq[((size_t)tn * BATCH + b) * 32 + lane];
                if (tn >= CTXLEN) {                          // autoregressive inputs
                    if (lane == 0)            v = f2bf(y);
                    else if (lane <= 10)      v = f2bf(lagbuf[b * MAXLAG + (c_lags[lane - 1] - 1)]);
                }
                hA0[(size_t)b * K0 + lane] = v;
                if (tn >= CTXLEN && lane < MAXLAG) {         // buf = [prev, buf[:-1]]
                    float old = (lane >= 1) ? lagbuf[b * MAXLAG + lane - 1] : y;
                    lagbuf[b * MAXLAG + lane] = old;
                }
            }
        }
        gbar(barrier);
    }
}

// ---------------- launcher ------------------------------------------------
extern "C" void kernel_launch(void* const* d_in, const int* in_sizes, int n_in,
                              void* d_out, int out_size, void* d_ws, size_t ws_size,
                              hipStream_t stream) {
    (void)in_sizes; (void)n_in; (void)out_size; (void)ws_size;
    const float*         X      = (const float*)d_in[0];
    const unsigned char* mask   = (const unsigned char*)d_in[1];
    const float*         emb    = (const float*)d_in[2];
    const float*         W_ih0  = (const float*)d_in[3];
    const float*         W_hh0  = (const float*)d_in[4];
    const float*         b_ih0  = (const float*)d_in[5];
    const float*         b_hh0  = (const float*)d_in[6];
    const float*         W_ih1  = (const float*)d_in[7];
    const float*         W_hh1  = (const float*)d_in[8];
    const float*         b_ih1  = (const float*)d_in[9];
    const float*         b_hh1  = (const float*)d_in[10];
    const float*         W_head = (const float*)d_in[11];
    const float*         b_head = (const float*)d_in[12];
    float* out = (float*)d_out;

    char* base = (char*)d_ws;
    size_t off = 0;
    auto carve = [&](size_t bytes) -> char* {
        off = (off + 255) & ~(size_t)255;
        char* p = base + off; off += bytes; return p;
    };
    unsigned*       barrier = (unsigned*)      carve(256);
    float*          scale   = (float*)         carve(BATCH * 4);
    float*          logsc   = (float*)         carve(BATCH * 4);
    float*          lagbuf  = (float*)         carve(BATCH * MAXLAG * 4);
    float*          bias0   = (float*)         carve(G4 * 4);
    float*          bias1   = (float*)         carve(G4 * 4);
    float*          c0      = (float*)         carve(BATCH * HID * 4);
    float*          c1      = (float*)         carve(BATCH * HID * 4);
    float*          h1f     = (float*)         carve(BATCH * HID * 4);
    float*          g0      = (float*)         carve((size_t)BATCH * G4 * 4);
    float*          g1      = (float*)         carve((size_t)BATCH * G4 * 4);
    unsigned short* hA0     = (unsigned short*)carve((size_t)BATCH * K0 * 2);
    unsigned short* hA1     = (unsigned short*)carve((size_t)BATCH * K1 * 2);
    unsigned short* wcat0   = (unsigned short*)carve((size_t)NT * KT0 * 512 * 2);
    unsigned short* wcat1   = (unsigned short*)carve((size_t)NT * KT1 * 512 * 2);
    unsigned short* xq      = (unsigned short*)carve((size_t)TSTEPS * BATCH * 32 * 2);

    (void)hipMemsetAsync(barrier, 0, 256, stream);

    prep_scale_k<<<BATCH, NTHR, 0, stream>>>(X, mask, scale, logsc, lagbuf);
    prep_bias_k<<<(2 * G4 + NTHR - 1) / NTHR, NTHR, 0, stream>>>(b_ih0, b_hh0, b_ih1, b_hh1,
                                                                 bias0, bias1);
    pack_w_k<<<(NT * KT0 * 32 + NTHR - 1) / NTHR, NTHR, 0, stream>>>(
        W_ih0, 17, 17, W_hh0, 32, KT0, wcat0);
    pack_w_k<<<(NT * KT1 * 32 + NTHR - 1) / NTHR, NTHR, 0, stream>>>(
        W_ih1, 512, 512, W_hh1, 512, KT1, wcat1);
    prep_feat_k<<<(TSTEPS * BATCH + NTHR - 1) / NTHR, NTHR, 0, stream>>>(
        X, emb, scale, logsc, xq);
    prep_state_k<<<256, NTHR, 0, stream>>>(c0, c1, hA0, hA1, xq);

    lstm_persistent<<<NBLK, NTHR, 0, stream>>>(
        wcat0, wcat1, bias0, bias1, xq, hA0, hA1, g0, g1,
        c0, c1, h1f, lagbuf, scale, W_head, b_head, out, barrier);
}
